// NODEDynamics_50036368998565
// MI455X (gfx1250) — compile-verified, matrix operand order checked by reading, not run on hardware
//
#include <hip/hip_runtime.h>
#include <hip/hip_bf16.h>
#include <stdint.h>

// ---------------------------------------------------------------------------
// GCN NODE dynamics for MI455X (gfx1250, wave32).
//  - GEMMs via v_wmma_f32_16x16x32_bf16 (bf16 operands, f32 accumulate)
//  - Edge aggregation: one wave32 per edge (32 lanes x float4 = 128 feats),
//    fp32 global atomics (fits in 192MB L2 -> on-die traffic)
//  - BatchNorm: channel-parallel reduction + fused norm/ReLU/bf16-convert
// ---------------------------------------------------------------------------

typedef __bf16 bf16_t;
typedef __attribute__((ext_vector_type(16))) bf16_t v16bf;
typedef __attribute__((ext_vector_type(8)))  float  v8f;

static __device__ __forceinline__ unsigned short f32_to_bf16_bits(float f) {
  union { float f; uint32_t u; } x; x.f = f;
  uint32_t u = x.u;
  uint32_t r = u + 0x7FFFu + ((u >> 16) & 1u);   // round-to-nearest-even
  return (unsigned short)(r >> 16);
}

// ------------------------------ small utility kernels ----------------------

__global__ void fill_k(float* __restrict__ p, float v, int n) {
  int i = blockIdx.x * blockDim.x + threadIdx.x;
  if (i < n) p[i] = v;
}

__global__ void zero_k(float* __restrict__ p, int n) {
  int i = blockIdx.x * blockDim.x + threadIdx.x;
  if (i < n) p[i] = 0.0f;
}

__global__ void deg_k(const int* __restrict__ col, const float* __restrict__ ew,
                      float* __restrict__ deg, int E) {
  int e = blockIdx.x * blockDim.x + threadIdx.x;
  if (e < E) atomicAdd(&deg[col[e]], ew[e]);
}

__global__ void dinv_k(const float* __restrict__ deg, float* __restrict__ dinv, int n) {
  int i = blockIdx.x * blockDim.x + threadIdx.x;
  if (i < n) {
    float d = deg[i];
    dinv[i] = (d > 0.0f) ? rsqrtf(d) : 0.0f;
  }
}

__global__ void norm_k(const int* __restrict__ row, const int* __restrict__ col,
                       const float* __restrict__ ew, const float* __restrict__ dinv,
                       float* __restrict__ nv, int E) {
  int e = blockIdx.x * blockDim.x + threadIdx.x;
  if (e < E) nv[e] = dinv[row[e]] * ew[e] * dinv[col[e]];
}

// W (K x Nout, row-major f32) -> Wb (Nout x K, row-major bf16)  [transpose]
__global__ void wprep_k(const float* __restrict__ W, unsigned short* __restrict__ Wb,
                        int K, int Nout) {
  int idx = blockIdx.x * blockDim.x + threadIdx.x;
  if (idx < K * Nout) {
    int n = idx / K;
    int k = idx - n * K;
    Wb[idx] = f32_to_bf16_bits(W[(size_t)k * Nout + n]);
  }
}

__global__ void tobf16_k(const float* __restrict__ x, unsigned short* __restrict__ xb,
                         long long n) {
  long long i = (long long)blockIdx.x * blockDim.x + threadIdx.x;
  if (i < n) xb[i] = f32_to_bf16_bits(x[i]);
}

// ------------------------------ WMMA GEMM ----------------------------------
// C[M x Nout] = A[M x K] * B^T (B stored as Nout x K), bf16 in / f32 out.
// One wave32 per 16x16 tile; K accumulated in steps of 32.

union FragU { v16bf v; uint4 q[2]; };

static __device__ __forceinline__ v16bf load_frag(const unsigned short* __restrict__ base,
                                                  int row, int K, int k0, int lane) {
  // ISA 16-bit A/B layout: lane's 16 elems = K runs [khalf, khalf+8) and
  // [khalf+16, khalf+24), khalf = 8*(lane>=16).  Two 16B contiguous loads.
  const int khalf = (lane >> 4) << 3;
  const unsigned short* p = base + (size_t)row * K + (k0 + khalf);
  FragU f;
  f.q[0] = *(const uint4*)(p);
  f.q[1] = *(const uint4*)(p + 16);
  return f.v;
}

__global__ void gemm_wmma_bf16(const unsigned short* __restrict__ Ag,
                               const unsigned short* __restrict__ Bg,
                               float* __restrict__ Cg,
                               int M, int K, int Nout) {
  const int lane = threadIdx.x & 31;
  const int wave = threadIdx.x >> 5;
  const int nT = Nout >> 4;
  const int totalTiles = (M >> 4) * nT;
  const int tile = blockIdx.x * (blockDim.x >> 5) + wave;
  if (tile >= totalTiles) return;            // wave-uniform: EXEC stays all-1s

  const int mT = tile / nT;
  const int nTile = tile - mT * nT;
  const int arow = (mT << 4) + (lane & 15);
  const int brow = (nTile << 4) + (lane & 15);

  v8f acc = {0.f, 0.f, 0.f, 0.f, 0.f, 0.f, 0.f, 0.f};
  for (int k0 = 0; k0 < K; k0 += 32) {
    v16bf a = load_frag(Ag, arow, K, k0, lane);
    v16bf b = load_frag(Bg, brow, K, k0, lane);
    acc = __builtin_amdgcn_wmma_f32_16x16x32_bf16(
        /*neg_a=*/false, a, /*neg_b=*/false, b,
        /*c_mod=*/(short)0, acc, /*reuse_a=*/false, /*reuse_b=*/false);
  }

  // D layout: lane -> col = nTile*16 + (lane&15); rows rbase..rbase+7
  const int ncol  = (nTile << 4) + (lane & 15);
  const int rbase = (mT << 4) + ((lane >> 4) << 3);
#pragma unroll
  for (int j = 0; j < 8; ++j)
    Cg[(size_t)(rbase + j) * Nout + ncol] = acc[j];
}

// ------------------------------ aggregation --------------------------------

// agg[i][c] = bias[c] + dinv[i]^2 * h[i][c]   (self-loop term + bias)
__global__ void agg_init_k(const float* __restrict__ h, const float* __restrict__ dinv,
                           const float* __restrict__ bias, float* __restrict__ agg,
                           int N, int H) {
  long long idx = (long long)blockIdx.x * blockDim.x + threadIdx.x;
  if (idx < (long long)N * H) {
    int c = (int)(idx % H);
    int i = (int)(idx / H);
    float di = dinv[i];
    agg[idx] = bias[c] + di * di * h[idx];
  }
}

// agg[col[e]][:] += norm[e] * h[row[e]][:]  -- (H/4) lanes per edge, float4
__global__ void agg_edges_k(const float* __restrict__ h, const int* __restrict__ row,
                            const int* __restrict__ col, const float* __restrict__ nv,
                            float* __restrict__ agg, int E, int H) {
  const int CH = H >> 2;
  long long t = (long long)blockIdx.x * blockDim.x + threadIdx.x;
  if (t >= (long long)E * CH) return;
  int e = (int)(t / CH);
  int c = (int)(t - (long long)e * CH) << 2;
  float w = nv[e];
  int r = row[e];
  int d = col[e];
  const float4 v = *(const float4*)(h + (size_t)r * H + c);
  float* out = agg + (size_t)d * H + c;
  atomicAdd(out + 0, w * v.x);
  atomicAdd(out + 1, w * v.y);
  atomicAdd(out + 2, w * v.z);
  atomicAdd(out + 3, w * v.w);
}

// ------------------------------ batchnorm ----------------------------------

__global__ void bn_stats_k(const float* __restrict__ a, float* __restrict__ stats,
                           int N, int H) {
  int c = threadIdx.x;                       // blockDim.x == H
  float s = 0.f, s2 = 0.f;
  for (int i = blockIdx.x; i < N; i += gridDim.x) {
    float v = a[(size_t)i * H + c];
    s += v;
    s2 += v * v;
  }
  atomicAdd(&stats[c], s);
  atomicAdd(&stats[H + c], s2);
}

__global__ void bn_final_k(float* __restrict__ stats, const float* __restrict__ g,
                           const float* __restrict__ be, int N, int H) {
  int c = threadIdx.x;
  if (c < H) {
    float inv_n = 1.0f / (float)N;
    float mu = stats[c] * inv_n;
    float var = stats[H + c] * inv_n - mu * mu;
    float rs = rsqrtf(var + 1e-5f);
    float sc = g[c] * rs;
    stats[2 * H + c] = sc;                   // scale
    stats[3 * H + c] = be[c] - mu * sc;      // shift
  }
}

__global__ void bn_relu_bf16_k(const float* __restrict__ a, const float* __restrict__ stats,
                               unsigned short* __restrict__ o, int N, int H) {
  long long idx = (long long)blockIdx.x * blockDim.x + threadIdx.x;
  if (idx < (long long)N * H) {
    int c = (int)(idx % H);
    float v = fmaf(a[idx], stats[2 * H + c], stats[3 * H + c]);
    o[idx] = f32_to_bf16_bits(fmaxf(v, 0.0f));
  }
}

// ------------------------------ launch -------------------------------------

static inline unsigned cdivu(long long a, long long b) {
  return (unsigned)((a + b - 1) / b);
}

extern "C" void kernel_launch(void* const* d_in, const int* in_sizes, int n_in,
                              void* d_out, int out_size, void* d_ws, size_t ws_size,
                              hipStream_t stream) {
  (void)n_in; (void)out_size; (void)ws_size;
  const float* x   = (const float*)d_in[1];
  const int*   ei  = (const int*)d_in[2];
  const float* ew  = (const float*)d_in[3];
  const float* W1  = (const float*)d_in[4];
  const float* b1  = (const float*)d_in[5];
  const float* g1  = (const float*)d_in[6];
  const float* be1 = (const float*)d_in[7];
  const float* W2  = (const float*)d_in[8];
  const float* b2  = (const float*)d_in[9];
  const float* g2  = (const float*)d_in[10];
  const float* be2 = (const float*)d_in[11];
  const float* Wf  = (const float*)d_in[12];
  const float* bfb = (const float*)d_in[13];

  const int D = in_sizes[13];          // 64
  const int H = in_sizes[5];           // 128
  const int N = in_sizes[1] / D;       // 100000
  const int E = in_sizes[2] / 2;       // 1600000
  const int* rowI = ei;
  const int* colI = ei + E;

  size_t off = 0;
  auto wsAlloc = [&](size_t bytes) -> void* {
    void* p = (void*)((char*)d_ws + off);
    off += (bytes + 255) & ~(size_t)255;
    return p;
  };
  float*          deg   = (float*)wsAlloc((size_t)N * 4);
  float*          dinv  = (float*)wsAlloc((size_t)N * 4);
  float*          nv    = (float*)wsAlloc((size_t)E * 4);
  unsigned short* Wb1   = (unsigned short*)wsAlloc((size_t)D * H * 2);
  unsigned short* Wb2   = (unsigned short*)wsAlloc((size_t)H * H * 2);
  unsigned short* Wbf   = (unsigned short*)wsAlloc((size_t)H * D * 2);
  unsigned short* Xb    = (unsigned short*)wsAlloc((size_t)N * D * 2);
  unsigned short* Hb    = (unsigned short*)wsAlloc((size_t)N * H * 2);
  float*          hf    = (float*)wsAlloc((size_t)N * H * 4);
  float*          aggf  = (float*)wsAlloc((size_t)N * H * 4);
  float*          stats = (float*)wsAlloc((size_t)4 * H * 4);
  float*          outp  = (float*)d_out;

  const int TB = 256;

  // --- graph normalization: deg (with self-loops), dinv, edge norms -------
  fill_k<<<cdivu(N, TB), TB, 0, stream>>>(deg, 1.0f, N);          // self-loop weight
  deg_k<<<cdivu(E, TB), TB, 0, stream>>>(colI, ew, deg, E);
  dinv_k<<<cdivu(N, TB), TB, 0, stream>>>(deg, dinv, N);
  norm_k<<<cdivu(E, TB), TB, 0, stream>>>(rowI, colI, ew, dinv, nv, E);

  // --- operand prep: bf16 weights (transposed to [Nout x K]) + bf16 x -----
  wprep_k<<<cdivu((long long)D * H, TB), TB, 0, stream>>>(W1, Wb1, D, H);
  wprep_k<<<cdivu((long long)H * H, TB), TB, 0, stream>>>(W2, Wb2, H, H);
  wprep_k<<<cdivu((long long)H * D, TB), TB, 0, stream>>>(Wf, Wbf, H, D);
  tobf16_k<<<cdivu((long long)N * D, TB), TB, 0, stream>>>(x, Xb, (long long)N * D);

  // --- layer 1: h = x @ W1 ; agg ; BN + ReLU -> bf16 ----------------------
  gemm_wmma_bf16<<<cdivu((long long)(N / 16) * (H / 16), 8), 256, 0, stream>>>(
      Xb, Wb1, hf, N, D, H);
  agg_init_k<<<cdivu((long long)N * H, TB), TB, 0, stream>>>(hf, dinv, b1, aggf, N, H);
  agg_edges_k<<<cdivu((long long)E * (H / 4), TB), TB, 0, stream>>>(
      hf, rowI, colI, nv, aggf, E, H);
  zero_k<<<1, 4 * H, 0, stream>>>(stats, 4 * H);
  bn_stats_k<<<512, H, 0, stream>>>(aggf, stats, N, H);
  bn_final_k<<<1, H, 0, stream>>>(stats, g1, be1, N, H);
  bn_relu_bf16_k<<<cdivu((long long)N * H, TB), TB, 0, stream>>>(aggf, stats, Hb, N, H);

  // --- layer 2: h = relu(bn(agg)) @ W2 ; agg ; BN + ReLU -> bf16 ----------
  gemm_wmma_bf16<<<cdivu((long long)(N / 16) * (H / 16), 8), 256, 0, stream>>>(
      Hb, Wb2, hf, N, H, H);
  agg_init_k<<<cdivu((long long)N * H, TB), TB, 0, stream>>>(hf, dinv, b2, aggf, N, H);
  agg_edges_k<<<cdivu((long long)E * (H / 4), TB), TB, 0, stream>>>(
      hf, rowI, colI, nv, aggf, E, H);
  zero_k<<<1, 4 * H, 0, stream>>>(stats, 4 * H);
  bn_stats_k<<<512, H, 0, stream>>>(aggf, stats, N, H);
  bn_final_k<<<1, H, 0, stream>>>(stats, g2, be2, N, H);
  bn_relu_bf16_k<<<cdivu((long long)N * H, TB), TB, 0, stream>>>(aggf, stats, Hb, N, H);

  // --- layer 3: dz/dt = agg(h @ Wf) + bf -> d_out -------------------------
  gemm_wmma_bf16<<<cdivu((long long)(N / 16) * (D / 16), 8), 256, 0, stream>>>(
      Hb, Wbf, hf, N, H, D);
  agg_init_k<<<cdivu((long long)N * D, TB), TB, 0, stream>>>(hf, dinv, bfb, outp, N, D);
  agg_edges_k<<<cdivu((long long)E * (D / 4), TB), TB, 0, stream>>>(
      hf, rowI, colI, nv, outp, E, D);
}
